// Memory_34248069218743
// MI455X (gfx1250) — matrix operand order repaired and checked by reading, not compiled, not run on hardware
//
#include <hip/hip_runtime.h>
#include <hip/hip_bf16.h>

typedef __attribute__((ext_vector_type(16))) __bf16 v16bf;
typedef __attribute__((ext_vector_type(8)))  float  v8f;

#define NROWS 65536   // B*H*W
#define DDIM  512
#define MSLOT 512
#define HWSZ  4096    // H*W
#define NEG_BIG (-3.0e38f)

// ---------------------------------------------------------------------------
// 0) zero q_update accumulator (must be re-zeroed every launch)
// ---------------------------------------------------------------------------
__global__ __launch_bounds__(256) void zero_qupdate(float* qupd) {
    int i = blockIdx.x * 256 + threadIdx.x;     // 262144 elements
    qupd[i] = 0.0f;
}

// ---------------------------------------------------------------------------
// 1) keys -> bf16 (row major) and bf16 transposed copies for WMMA B operands
// ---------------------------------------------------------------------------
__global__ __launch_bounds__(256) void prep_keys(const float* keys, __bf16* keysb, __bf16* keysTb) {
    int i = blockIdx.x * 256 + threadIdx.x;     // 262144
    int m = i >> 9, d = i & 511;
    float v = keys[i];
    keysb[i] = (__bf16)v;                       // [m][d]
    keysTb[(size_t)d * MSLOT + m] = (__bf16)v;  // [d][m]
}

// ---------------------------------------------------------------------------
// 2) L2-normalize query over D; result written to upd_q[:, :512] which is the
//    normalized query in the ORIGINAL (B,D,H,W) layout. Thread-per-row,
//    d-loop => coalesced across threads (consecutive hw).
// ---------------------------------------------------------------------------
__global__ __launch_bounds__(256) void normalize_q(const float* query, float* out_updq) {
    int r  = blockIdx.x * 256 + threadIdx.x;    // 65536 rows
    int b  = r >> 12, hw = r & 4095;
    const float* src = query + (size_t)b * (DDIM * HWSZ) + hw;
    float ss = 0.0f;
    for (int d = 0; d < DDIM; ++d) { float q = src[(size_t)d * HWSZ]; ss += q * q; }
    float inv = 1.0f / fmaxf(sqrtf(ss), 1e-12f);
    float* dst = out_updq + (size_t)b * (2 * DDIM * HWSZ) + hw;
    for (int d = 0; d < DDIM; ++d) dst[(size_t)d * HWSZ] = src[(size_t)d * HWSZ] * inv;
}

// ---------------------------------------------------------------------------
// 3) GEMM1: score[N,512] = qr[N,512] @ keys^T.  bf16 WMMA, f32 accumulate.
//    One block = one 16-row strip; 8 waves x 4 col-tiles = all 512 columns.
//    A staged in LDS from the (B,D,H,W)-layout normalized query.
// ---------------------------------------------------------------------------
__global__ __launch_bounds__(256) void gemm_score(const float* qnorm, const __bf16* keysb,
                                                  float* score) {
    __shared__ __bf16 Atile[16][DDIM + 2];      // +2 bf16 pad -> 257-word row stride
    const int r0  = blockIdx.x * 16;
    const int b   = r0 >> 12;
    const int hw0 = r0 & 4095;
    const int t   = threadIdx.x;
    {   // stage A (strided source, 64B-coalesced), convert to bf16
        const int x = t & 15;                   // row (hw offset / M of tile)
        const int y = t >> 4;                   // 16 d-lanes
        const float* base = qnorm + (size_t)b * (2 * DDIM * HWSZ) + hw0 + x;
        for (int dd = 0; dd < DDIM / 16; ++dd) {
            int d = y + dd * 16;
            Atile[x][d] = (__bf16)base[(size_t)d * HWSZ];
        }
    }
    __syncthreads();
    const int wave = t >> 5, lane = t & 31;
    const int m = lane & 15, g = lane >> 4;
    v8f acc[4] = {};
    for (int kt = 0; kt < DDIM / 32; ++kt) {
        const int K0 = kt * 32;
        v16bf a{};
#pragma unroll
        for (int v = 0; v < 8; ++v) {           // A layout: k = g*16 + v*2 + p
            int kb = K0 + g * 16 + v * 2;
            a[2 * v]     = Atile[m][kb];
            a[2 * v + 1] = Atile[m][kb + 1];
        }
#pragma unroll
        for (int j = 0; j < 4; ++j) {
            const int C0 = (wave * 4 + j) * 16;
            // B[k,n] = keys[C0+n][K0+k]; per lane n=m fixed, k contiguous in keysb row
            const __bf16* brow = keysb + (size_t)(C0 + m) * DDIM + K0 + g * 16;
            v16bf bf{};
#pragma unroll
            for (int v = 0; v < 8; ++v) { bf[2 * v] = brow[2 * v]; bf[2 * v + 1] = brow[2 * v + 1]; }
            acc[j] = __builtin_amdgcn_wmma_f32_16x16x32_bf16(false, a, false, bf,
                                                             (short)0, acc[j], false, false);
        }
    }
#pragma unroll
    for (int j = 0; j < 4; ++j) {               // C/D: m = rr + 8*g, n = lane&15
        const int C0 = (wave * 4 + j) * 16;
        const int mb = g * 8;
#pragma unroll
        for (int rr = 0; rr < 8; ++rr)
            score[(size_t)(r0 + mb + rr) * MSLOT + C0 + m] = acc[j][rr];
    }
}

// ---------------------------------------------------------------------------
// 4) Row softmax (s_memory) + top-2 indices per row
// ---------------------------------------------------------------------------
__global__ __launch_bounds__(128) void row_softmax_top2(const float* score, float* smem_out,
                                                        int* idx1, int* idx2) {
    const int r = blockIdx.x, t = threadIdx.x;
    const size_t base = (size_t)r * MSLOT;
    float v[4];
    float t1 = NEG_BIG, t2 = NEG_BIG; int j1 = 0, j2 = 0;
#pragma unroll
    for (int i = 0; i < 4; ++i) {
        int d = t + 128 * i;
        v[i] = score[base + d];
        if (v[i] > t1) { t2 = t1; j2 = j1; t1 = v[i]; j1 = d; }
        else if (v[i] > t2) { t2 = v[i]; j2 = d; }
    }
    __shared__ float sv1[128], sv2[128];
    __shared__ int   si1[128], si2[128];
    sv1[t] = t1; si1[t] = j1; sv2[t] = t2; si2[t] = j2;
    __syncthreads();
    for (int off = 64; off > 0; off >>= 1) {
        if (t < off) {
            float w1 = sv1[t + off]; int k1 = si1[t + off];
            float w2 = sv2[t + off]; int k2 = si2[t + off];
            float a1 = sv1[t]; int b1 = si1[t]; float a2 = sv2[t]; int b2 = si2[t];
            if (w1 > a1) { a2 = a1; b2 = b1; a1 = w1; b1 = k1; if (w2 > a2) { a2 = w2; b2 = k2; } }
            else if (w1 > a2) { a2 = w1; b2 = k1; }
            sv1[t] = a1; si1[t] = b1; sv2[t] = a2; si2[t] = b2;
        }
        __syncthreads();
    }
    const float rmax = sv1[0]; const int gi1 = si1[0], gi2 = si2[0];
    __syncthreads();
    float ev[4], part = 0.0f;
#pragma unroll
    for (int i = 0; i < 4; ++i) { ev[i] = __expf(v[i] - rmax); part += ev[i]; }
    sv1[t] = part;
    __syncthreads();
    for (int off = 64; off > 0; off >>= 1) { if (t < off) sv1[t] += sv1[t + off]; __syncthreads(); }
    const float inv = 1.0f / sv1[0];
#pragma unroll
    for (int i = 0; i < 4; ++i) smem_out[base + t + 128 * i] = ev[i] * inv;
    if (t == 0) { idx1[r] = gi1; idx2[r] = gi2; }
}

// ---------------------------------------------------------------------------
// 5) Column softmax stage 1: online (max,sum) partials per 256-row chunk
// ---------------------------------------------------------------------------
__global__ __launch_bounds__(256) void col_pass1(const float* score, float* pm, float* ps) {
    const int x = threadIdx.x & 31;             // column offset (coalesced)
    const int y = threadIdx.x >> 5;             // 8 row lanes
    const int c = blockIdx.x * 32 + x;
    const int rbase = blockIdx.y * 256;
    float m = NEG_BIG, s = 0.0f;
    for (int i = 0; i < 32; ++i) {
        float val = score[(size_t)(rbase + y + 8 * i) * MSLOT + c];
        float mn = fmaxf(m, val);
        s = s * __expf(m - mn) + __expf(val - mn);
        m = mn;
    }
    __shared__ float lm[8][32], ls[8][32];
    lm[y][x] = m; ls[y][x] = s;
    __syncthreads();
    for (int off = 4; off > 0; off >>= 1) {
        if (y < off) {
            float m2 = lm[y + off][x], s2 = ls[y + off][x];
            float mn = fmaxf(lm[y][x], m2);
            ls[y][x] = ls[y][x] * __expf(lm[y][x] - mn) + s2 * __expf(m2 - mn);
            lm[y][x] = mn;
        }
        __syncthreads();
    }
    if (y == 0) { pm[(size_t)blockIdx.y * MSLOT + c] = lm[0][x]; ps[(size_t)blockIdx.y * MSLOT + c] = ls[0][x]; }
}

// 5b) stage 2: reduce 256 chunk-partials per column
__global__ __launch_bounds__(256) void col_pass2(const float* pm, const float* ps,
                                                 float* colmax, float* colsum) {
    const int c = blockIdx.x, t = threadIdx.x;
    __shared__ float lm[256], ls[256];
    lm[t] = pm[(size_t)t * MSLOT + c]; ls[t] = ps[(size_t)t * MSLOT + c];
    __syncthreads();
    for (int off = 128; off > 0; off >>= 1) {
        if (t < off) {
            float m2 = lm[t + off], s2 = ls[t + off];
            float mn = fmaxf(lm[t], m2);
            ls[t] = ls[t] * __expf(lm[t] - mn) + s2 * __expf(m2 - mn);
            lm[t] = mn;
        }
        __syncthreads();
    }
    if (t == 0) { colmax[c] = lm[0]; colsum[c] = ls[0]; }
}

// 5c) s_query elementwise write
__global__ __launch_bounds__(256) void squery_write(const float* score, const float* colmax,
                                                    const float* colsum, float* sq) {
    size_t i = (size_t)blockIdx.x * 256 + threadIdx.x;   // 33554432
    int c = (int)(i & 511);
    sq[i] = __expf(score[i] - colmax[c]) / colsum[c];
}

// ---------------------------------------------------------------------------
// 6) GEMM2: upd_q[:,512:] = s_memory @ keys, stored directly into the
//    transposed (B,2D,H,W) output layout. A read from f32 s_memory (fp32 in
//    d_out), converted to bf16 during LDS staging.
// ---------------------------------------------------------------------------
__global__ __launch_bounds__(256) void gemm_updq(const float* smem, const __bf16* keysTb,
                                                 float* out_updq) {
    __shared__ __bf16 Atile[16][DDIM + 2];
    const int r0  = blockIdx.x * 16;
    const int b   = r0 >> 12;
    const int hw0 = r0 & 4095;
    const int t   = threadIdx.x;
    for (int it = 0; it < 32; ++it) {           // linear coalesced staging
        int idx = t + 256 * it;                 // over 16*512 elements
        int row = idx >> 9, d = idx & 511;
        Atile[row][d] = (__bf16)smem[(size_t)(r0 + row) * MSLOT + d];
    }
    __syncthreads();
    const int wave = t >> 5, lane = t & 31;
    const int m = lane & 15, g = lane >> 4;
    v8f acc[4] = {};
    for (int kt = 0; kt < MSLOT / 32; ++kt) {
        const int K0 = kt * 32;
        v16bf a{};
#pragma unroll
        for (int v = 0; v < 8; ++v) {
            int kb = K0 + g * 16 + v * 2;
            a[2 * v]     = Atile[m][kb];
            a[2 * v + 1] = Atile[m][kb + 1];
        }
#pragma unroll
        for (int j = 0; j < 4; ++j) {
            const int C0 = (wave * 4 + j) * 16;
            // B[k,n] = keys[k][C0+n] = keysT[C0+n][k], contiguous in k
            const __bf16* brow = keysTb + (size_t)(C0 + m) * MSLOT + K0 + g * 16;
            v16bf bf{};
#pragma unroll
            for (int v = 0; v < 8; ++v) { bf[2 * v] = brow[2 * v]; bf[2 * v + 1] = brow[2 * v + 1]; }
            acc[j] = __builtin_amdgcn_wmma_f32_16x16x32_bf16(false, a, false, bf,
                                                             (short)0, acc[j], false, false);
        }
    }
#pragma unroll
    for (int j = 0; j < 4; ++j) {
        const int C0 = (wave * 4 + j) * 16;
        const int mb = g * 8;
        float* dst = out_updq + (size_t)b * (2 * DDIM * HWSZ)
                   + (size_t)(DDIM + C0 + m) * HWSZ + hw0 + mb;
#pragma unroll
        for (int rr = 0; rr < 8; ++rr) dst[rr] = acc[j][rr];
    }
}

// ---------------------------------------------------------------------------
// 7) segment-sum scatter: q_update[g] += wgt[r] * qr[r,:]
//    wgt[r] = exp(score[r,g] - colmax[g])  (== (s_query/colmax_sq) at g)
// ---------------------------------------------------------------------------
__global__ __launch_bounds__(256) void scatter_qupdate(const float* qnorm, const float* score,
                                                       const float* colmax, const int* idx1,
                                                       float* qupd) {
    int r = blockIdx.x * 256 + threadIdx.x;
    int g = idx1[r];
    float w = __expf(score[(size_t)r * MSLOT + g] - colmax[g]);
    int b = r >> 12, hw = r & 4095;
    const float* qb = qnorm + (size_t)b * (2 * DDIM * HWSZ) + hw;
    float* dst = qupd + (size_t)g * DDIM;
    for (int d = 0; d < DDIM; ++d)
        __hip_atomic_fetch_add(dst + d, w * qb[(size_t)d * HWSZ],
                               __ATOMIC_RELAXED, __HIP_MEMORY_SCOPE_AGENT);
}

// ---------------------------------------------------------------------------
// 8) compactness + separateness (block partials, deterministic final reduce)
// ---------------------------------------------------------------------------
__global__ __launch_bounds__(256) void compact_sep(const float* qnorm, const float* keys,
                                                   const int* idx1, const int* idx2,
                                                   float* cpt_part, float* sep_part) {
    int r = blockIdx.x * 256 + threadIdx.x;
    int g1 = idx1[r], g2 = idx2[r];
    int b = r >> 12, hw = r & 4095;
    const float* qb = qnorm + (size_t)b * (2 * DDIM * HWSZ) + hw;
    const float* k1 = keys + (size_t)g1 * DDIM;
    const float* k2 = keys + (size_t)g2 * DDIM;
    float cpt = 0.0f, dp = 0.0f, dn = 0.0f;
    for (int d = 0; d < DDIM; ++d) {
        float q  = qb[(size_t)d * HWSZ];
        float e1 = q - k1[d];           cpt += e1 * e1;
        float ep = e1 + 1e-6f;          dp  += ep * ep;
        float en = q - k2[d] + 1e-6f;   dn  += en * en;
    }
    float sep = fmaxf(sqrtf(dp) - sqrtf(dn) + 1.0f, 0.0f);
    __shared__ float rc[256], rs[256];
    int t = threadIdx.x;
    rc[t] = cpt; rs[t] = sep;
    __syncthreads();
    for (int off = 128; off > 0; off >>= 1) {
        if (t < off) { rc[t] += rc[t + off]; rs[t] += rs[t + off]; }
        __syncthreads();
    }
    if (t == 0) { cpt_part[blockIdx.x] = rc[0]; sep_part[blockIdx.x] = rs[0]; }
}

__global__ __launch_bounds__(256) void reduce_scalars(const float* cpt_part, const float* sep_part,
                                                      float* out_scalars) {
    __shared__ float a[256], s[256];
    int t = threadIdx.x;
    a[t] = cpt_part[t]; s[t] = sep_part[t];
    __syncthreads();
    for (int off = 128; off > 0; off >>= 1) {
        if (t < off) { a[t] += a[t + off]; s[t] += s[t + off]; }
        __syncthreads();
    }
    if (t == 0) {
        out_scalars[0] = s[0] / (float)NROWS;                   // separateness
        out_scalars[1] = a[0] / ((float)NROWS * (float)DDIM);   // compactness
    }
}

// ---------------------------------------------------------------------------
// 9) upd_mem = l2norm(q_update + keys)
// ---------------------------------------------------------------------------
__global__ __launch_bounds__(128) void upd_mem_kernel(const float* qupd, const float* keys,
                                                      float* out_um) {
    const int mrow = blockIdx.x, t = threadIdx.x;
    const size_t base = (size_t)mrow * DDIM;
    float v[4]; float ss = 0.0f;
#pragma unroll
    for (int i = 0; i < 4; ++i) {
        int d = t + 128 * i;
        v[i] = qupd[base + d] + keys[base + d];
        ss += v[i] * v[i];
    }
    __shared__ float red[128];
    red[t] = ss;
    __syncthreads();
    for (int off = 64; off > 0; off >>= 1) { if (t < off) red[t] += red[t + off]; __syncthreads(); }
    float inv = 1.0f / fmaxf(sqrtf(red[0]), 1e-12f);
#pragma unroll
    for (int i = 0; i < 4; ++i) out_um[base + t + 128 * i] = v[i] * inv;
}

// ---------------------------------------------------------------------------
extern "C" void kernel_launch(void* const* d_in, const int* in_sizes, int n_in,
                              void* d_out, int out_size, void* d_ws, size_t ws_size,
                              hipStream_t stream) {
    (void)in_sizes; (void)n_in; (void)out_size; (void)ws_size;
    const float* query = (const float*)d_in[0];
    const float* keys  = (const float*)d_in[1];

    // output slices (all f32, concatenated in return order)
    float* out_updq    = (float*)d_out;                       // 16*1024*64*64
    float* out_updmem  = out_updq + (size_t)67108864;         // 512*512
    float* out_squery  = out_updmem + (size_t)262144;         // 65536*512
    float* out_smem    = out_squery + (size_t)33554432;       // 65536*512
    float* out_scalars = out_smem   + (size_t)33554432;       // sep, cpt

    // workspace layout (256B aligned slabs)
    char* w = (char*)d_ws;
    auto take = [&](size_t bytes) { char* p = w; w += (bytes + 255) & ~(size_t)255; return p; };
    float*  score    = (float*) take((size_t)NROWS * MSLOT * 4);   // 134 MB
    __bf16* keysb    = (__bf16*)take((size_t)MSLOT * DDIM * 2);
    __bf16* keysTb   = (__bf16*)take((size_t)DDIM * MSLOT * 2);
    float*  pm       = (float*) take((size_t)256 * MSLOT * 4);
    float*  ps       = (float*) take((size_t)256 * MSLOT * 4);
    float*  colmax   = (float*) take((size_t)MSLOT * 4);
    float*  colsum   = (float*) take((size_t)MSLOT * 4);
    int*    idx1     = (int*)   take((size_t)NROWS * 4);
    int*    idx2     = (int*)   take((size_t)NROWS * 4);
    float*  qupd     = (float*) take((size_t)MSLOT * DDIM * 4);
    float*  cpt_part = (float*) take((size_t)256 * 4);
    float*  sep_part = (float*) take((size_t)256 * 4);

    zero_qupdate   <<<1024, 256, 0, stream>>>(qupd);
    prep_keys      <<<1024, 256, 0, stream>>>(keys, keysb, keysTb);
    normalize_q    <<<256, 256, 0, stream>>>(query, out_updq);
    gemm_score     <<<4096, 256, 0, stream>>>(out_updq, keysb, score);
    row_softmax_top2<<<65536, 128, 0, stream>>>(score, out_smem, idx1, idx2);
    col_pass1      <<<dim3(16, 256), 256, 0, stream>>>(score, pm, ps);
    col_pass2      <<<512, 256, 0, stream>>>(pm, ps, colmax, colsum);
    squery_write   <<<131072, 256, 0, stream>>>(score, colmax, colsum, out_squery);
    gemm_updq      <<<4096, 256, 0, stream>>>(out_smem, keysTb, out_updq);
    scatter_qupdate<<<256, 256, 0, stream>>>(out_updq, score, colmax, idx1, qupd);
    compact_sep    <<<256, 256, 0, stream>>>(out_updq, keys, idx1, idx2, cpt_part, sep_part);
    reduce_scalars <<<1, 256, 0, stream>>>(cpt_part, sep_part, out_scalars);
    upd_mem_kernel <<<512, 128, 0, stream>>>(qupd, keys, out_updmem);
}
